// RnnInferenceNetwork_11278584120103
// MI455X (gfx1250) — compile-verified, matrix operand order checked by reading, not run on hardware
//
#include <hip/hip_runtime.h>
#include <hip/hip_bf16.h>

// ---------------------------------------------------------------------------
// Variational LSTM inference net on MI455X (gfx1250, wave32, WMMA).
//  - persistent time loop: 64 blocks x 16 rows of N=1024; recurrence state
//    (h, z) stays in LDS, c stays in WMMA accumulators for all 100 steps.
//  - gate GEMM: v_wmma_f32_16x16x32_bf16, fused K=832 (x|z|h), weights
//    repacked to bf16 once (3.5 MB -> resident in 192MB L2).
//  - x_t double-buffered in LDS via gfx1250 async LDS loads
//    (global_load_async_to_lds_b128 + s_wait_asynccnt), issued one step
//    ahead so the fetch overlaps the gate GEMM.
//  - gate biases folded into accumulator init (splat), not a post-pass.
// ---------------------------------------------------------------------------

typedef __attribute__((ext_vector_type(16))) __bf16 v16bf;
typedef __attribute__((ext_vector_type(8)))  float  v8f;

#define OBS   256
#define HID   512
#define LAT   64
#define BB    256
#define TT    100
#define SS    4
#define NN    (BB * SS)            // 1024
#define KTOT  (OBS + LAT + HID)    // 832 fused K for the gate GEMM
#define G4H   (4 * HID)            // 2048 gate columns

#define WG_ELEMS ((size_t)G4H * KTOT)        // fused gate weights, bf16
#define WL_ELEMS ((size_t)(2 * LAT) * HID)   // [W_mu ; W_sig], bf16

union FragU { v16bf v; uint4 u[2]; };

// Per-lane WMMA 16-bit operand fragment: elements {k..k+7} and {k+16..k+23}
// (caller folds in the lane's K-offset). Two 16B vector loads.
__device__ __forceinline__ v16bf load_frag(const __bf16* p) {
  FragU f;
  f.u[0] = *(const uint4*)(p);
  f.u[1] = *(const uint4*)(p + 16);
  return f.v;
}

__device__ __forceinline__ float sigm(float x) { return 1.0f / (1.0f + __expf(-x)); }
__device__ __forceinline__ float softplusf(float x) {
  return fmaxf(x, 0.0f) + log1pf(__expf(-fabsf(x)));
}

// gfx1250 async global->LDS copy (ASYNCcnt): vdst = LDS byte address,
// GVS addressing: s[base64] + voffset32.
__device__ __forceinline__ void async_g2l_b128(uint32_t lds_addr, uint32_t voff,
                                               const void* base) {
  asm volatile("global_load_async_to_lds_b128 %0, %1, %2"
               :: "v"(lds_addr), "v"(voff), "s"(base) : "memory");
}
__device__ __forceinline__ void wait_async0() {
  asm volatile("s_wait_asynccnt 0x0" ::: "memory");
}

// ---------------------------------------------------------------------------
// Prep: pack weights to bf16, fuse biases.
//   Wg[r][k] : k<320 -> W_ih[r][k], else W_hh[r][k-320]   (2048 x 832)
//   Wl[r][k] : r<64 -> W_mu[r][k], else W_sig[r-64][k]    (128 x 512)
//   bias[c]  = b_ih[c] + b_hh[c]
// ---------------------------------------------------------------------------
__global__ void vrnn_prep_kernel(const float* __restrict__ W_ih,
                                 const float* __restrict__ W_hh,
                                 const float* __restrict__ b_ih,
                                 const float* __restrict__ b_hh,
                                 const float* __restrict__ W_mu,
                                 const float* __restrict__ W_sig,
                                 __bf16* __restrict__ Wg,
                                 __bf16* __restrict__ Wl,
                                 float* __restrict__ bias) {
  const int stride = gridDim.x * blockDim.x;
  const int tid0 = blockIdx.x * blockDim.x + threadIdx.x;
  for (size_t i = tid0; i < WG_ELEMS; i += stride) {
    int r = (int)(i / KTOT), k = (int)(i % KTOT);
    float v = (k < OBS + LAT) ? W_ih[(size_t)r * (OBS + LAT) + k]
                              : W_hh[(size_t)r * HID + (k - OBS - LAT)];
    Wg[i] = (__bf16)v;
  }
  for (size_t i = tid0; i < WL_ELEMS; i += stride) {
    int r = (int)(i / HID), k = (int)(i % HID);
    float v = (r < LAT) ? W_mu[(size_t)r * HID + k] : W_sig[(size_t)(r - LAT) * HID + k];
    Wl[i] = (__bf16)v;
  }
  for (int i = tid0; i < G4H; i += stride) bias[i] = b_ih[i] + b_hh[i];
}

// ---------------------------------------------------------------------------
// Main persistent kernel: grid = 64 blocks (16 rows each), 512 threads = 16 waves.
// ---------------------------------------------------------------------------
__global__ __launch_bounds__(512)
void vrnn_kernel(const float* __restrict__ inputs,   // [B,T,OBS]
                 const float* __restrict__ eps,      // [T,N,LAT]
                 const float* __restrict__ b_mu,
                 const float* __restrict__ b_sig,
                 const __bf16* __restrict__ Wg,
                 const __bf16* __restrict__ Wl,
                 const float* __restrict__ bias,
                 float* __restrict__ out) {
  // Fused A operand for the gate GEMM: [16 rows][ x(256) | z(64) | h(512) ] bf16.
  __shared__ __bf16 A[16][KTOT];
  __shared__ float  xstage[2][16][OBS];   // double-buffered raw f32 x tiles
  __shared__ float  rowacc_lp[16];
  __shared__ float  rowacc_ent[16];

  const int tid   = threadIdx.x;
  const int wave  = tid >> 5;
  const int lane  = tid & 31;
  const int l16   = lane & 15;
  const int khalf = lane >> 4;     // lanes 16..31 take the +8 K-offset half
  const int koff  = khalf * 8;
  const int rowBase = blockIdx.x * 16;

  float* out_s   = out;                                // sampled_s : n*T*L + t*L + l
  float* out_ent = out + (size_t)NN * TT * LAT;        // entropies : n*T + t
  float* out_lp  = out_ent + (size_t)NN * TT;          // log_probs : n*T + t
  float* out_c   = out_lp + (size_t)NN * TT;           // rnn_states: n*H*T + h*T + t

  const int hbase = wave * 32;     // 32 hidden columns owned by this wave

  // This thread's two 16B chunks of the 16x256 x-tile (load & convert the
  // same slice -> wave-local s_wait_asynccnt is the only sync needed).
  const int ch0 = tid, ch1 = tid + 512;          // of 1024 chunks
  const int xr0 = ch0 >> 6, xc0 = (ch0 & 63) << 2;
  const int xr1 = ch1 >> 6, xc1 = (ch1 & 63) << 2;
  const uint32_t grow0 = (uint32_t)((rowBase + xr0) & (BB - 1)) * TT;
  const uint32_t grow1 = (uint32_t)((rowBase + xr1) & (BB - 1)) * TT;
  uint32_t xs_addr[2];
  xs_addr[0] = (uint32_t)(uintptr_t)&xstage[0][0][0];
  xs_addr[1] = (uint32_t)(uintptr_t)&xstage[1][0][0];

  // kick off async stage of x_0 while we zero the recurrence state
  async_g2l_b128(xs_addr[0] + (uint32_t)(xr0 * OBS + xc0) * 4u,
                 ((grow0 + 0) * OBS + xc0) * 4u, inputs);
  async_g2l_b128(xs_addr[0] + (uint32_t)(xr1 * OBS + xc1) * 4u,
                 ((grow1 + 0) * OBS + xc1) * 4u, inputs);

  // zero z|h region of A (h0 = z0 = 0)
  for (int i = tid; i < 16 * (LAT + HID); i += 512) {
    int r = i / (LAT + HID), c = i % (LAT + HID);
    A[r][OBS + c] = (__bf16)0.0f;
  }

  v8f cst[2] = {};   // persistent c-state: 2 col-tiles x 8 rows per lane

  // B row pointers + fused biases for this wave's 8 gate tiles (i,f,g,o x 2).
  const __bf16* browp[8];
  float bgate[8];
#pragma unroll
  for (int g = 0; g < 4; ++g)
#pragma unroll
    for (int h2 = 0; h2 < 2; ++h2) {
      int col = g * HID + hbase + h2 * 16 + l16;       // B col = lane&15 within tile
      browp[g * 2 + h2] = Wg + (size_t)col * KTOT + koff;
      bgate[g * 2 + h2] = bias[col];
    }
  const int  latc = (wave & 3) * 16 + l16;             // latent col for waves 0..3
  const float bm_s = b_mu[latc];
  const float bs_s = b_sig[latc];

  const float LOG2PI = 1.8378770664093453f;

  for (int t = 0; t < TT; ++t) {
    // ---- phase A: staged x_t -> A (f32 -> bf16); prefetch x_{t+1} async ----
    wait_async0();
    {
      float4 v0 = *(const float4*)&xstage[t & 1][xr0][xc0];
      float4 v1 = *(const float4*)&xstage[t & 1][xr1][xc1];
      A[xr0][xc0 + 0] = (__bf16)v0.x; A[xr0][xc0 + 1] = (__bf16)v0.y;
      A[xr0][xc0 + 2] = (__bf16)v0.z; A[xr0][xc0 + 3] = (__bf16)v0.w;
      A[xr1][xc1 + 0] = (__bf16)v1.x; A[xr1][xc1 + 1] = (__bf16)v1.y;
      A[xr1][xc1 + 2] = (__bf16)v1.z; A[xr1][xc1 + 3] = (__bf16)v1.w;
    }
    if (t + 1 < TT) {
      uint32_t buf = xs_addr[(t + 1) & 1];
      async_g2l_b128(buf + (uint32_t)(xr0 * OBS + xc0) * 4u,
                     ((grow0 + (uint32_t)(t + 1)) * OBS + xc0) * 4u, inputs);
      async_g2l_b128(buf + (uint32_t)(xr1 * OBS + xc1) * 4u,
                     ((grow1 + (uint32_t)(t + 1)) * OBS + xc1) * 4u, inputs);
    }
    __syncthreads();   // B1: x/z/h ready

    // ---- phase B: gate GEMM, 8 tiles x 26 K-steps of wmma bf16 ----
    v8f acc[8];
#pragma unroll
    for (int q = 0; q < 8; ++q) acc[q] = (v8f)bgate[q];   // bias folded into init
    {
      const __bf16* arow = &A[l16][koff];              // A row = lane&15 (ISA layout)
      for (int k0 = 0; k0 < KTOT; k0 += 32) {
        v16bf af = load_frag(arow + k0);
#pragma unroll
        for (int q = 0; q < 8; ++q) {
          v16bf bf = load_frag(browp[q] + k0);
          acc[q] = __builtin_amdgcn_wmma_f32_16x16x32_bf16(
              false, af, false, bf, (short)0, acc[q], false, false);
        }
      }
    }
    __syncthreads();   // B2: all LDS reads for this step's GEMM are done

    if (tid < 16) { rowacc_lp[tid] = 0.0f; rowacc_ent[tid] = 0.0f; }

    // ---- phase C: LSTM pointwise; h -> LDS, c -> rnn_states ----
#pragma unroll
    for (int h2 = 0; h2 < 2; ++h2) {
      int hcol = hbase + h2 * 16 + l16;
#pragma unroll
      for (int r = 0; r < 8; ++r) {
        int row = r + khalf * 8;                       // D row = vgpr + 8*(lane>=16)
        float iv = acc[0 + h2][r];
        float fv = acc[2 + h2][r];
        float gv = acc[4 + h2][r];
        float ov = acc[6 + h2][r];
        float cn = sigm(fv) * cst[h2][r] + sigm(iv) * tanhf(gv);
        float hn = sigm(ov) * tanhf(cn);
        cst[h2][r] = cn;
        A[row][OBS + LAT + hcol] = (__bf16)hn;
        int n = rowBase + row;
        out_c[(size_t)n * (HID * TT) + (size_t)hcol * TT + t] = cn;  // faithful dim scramble
      }
    }
    __syncthreads();   // B3: h ready in LDS, rowacc reset visible

    // ---- phase D: waves 0..3: mu/sig GEMM (16x16 tile each, K=512) + rsample ----
    if (wave < 4) {
      v8f amu = (v8f)bm_s, asg = (v8f)bs_s;            // biases folded into init
      const __bf16* arow   = &A[l16][OBS + LAT + koff];
      const __bf16* bmurow = Wl + (size_t)latc * HID + koff;
      const __bf16* bsgrow = Wl + (size_t)(LAT + latc) * HID + koff;
      for (int k0 = 0; k0 < HID; k0 += 32) {
        v16bf af = load_frag(arow + k0);
        v16bf bm = load_frag(bmurow + k0);
        v16bf bs = load_frag(bsgrow + k0);
        amu = __builtin_amdgcn_wmma_f32_16x16x32_bf16(false, af, false, bm, (short)0, amu, false, false);
        asg = __builtin_amdgcn_wmma_f32_16x16x32_bf16(false, af, false, bs, (short)0, asg, false, false);
      }
      float lp_r[8], en_r[8];
#pragma unroll
      for (int r = 0; r < 8; ++r) {
        int row = r + khalf * 8;
        int n = rowBase + row;
        float mu = amu[r];
        float sg = softplusf(asg[r]);
        float e  = eps[((size_t)t * NN + n) * LAT + latc];
        float z  = mu + sg * e;                        // rsample; (z-mu)/sigma == e
        float ls = __logf(sg);
        lp_r[r] = -0.5f * e * e - ls;
        en_r[r] = ls;
        out_s[(size_t)n * (TT * LAT) + t * LAT + latc] = z;
        A[row][OBS + latc] = (__bf16)z;                // feed next step's gate GEMM
      }
      // reduce over this wave's 16 latent columns (stays within 16-lane halves)
#pragma unroll
      for (int r = 0; r < 8; ++r) {
        float vlp = lp_r[r], ven = en_r[r];
#pragma unroll
        for (int m = 1; m < 16; m <<= 1) {
          vlp += __shfl_xor(vlp, m, 32);
          ven += __shfl_xor(ven, m, 32);
        }
        if (l16 == 0) {
          atomicAdd(&rowacc_lp[r + khalf * 8], vlp);   // ds_add_f32
          atomicAdd(&rowacc_ent[r + khalf * 8], ven);
        }
      }
    }
    __syncthreads();   // B4: row sums + new z complete

    // ---- phase E: finalize logp / entropy per row ----
    if (tid < 16) {
      int n = rowBase + tid;
      out_lp[(size_t)n * TT + t]  = rowacc_lp[tid] - 0.5f * LOG2PI * (float)LAT;
      out_ent[(size_t)n * TT + t] = rowacc_ent[tid] + (float)LAT * (0.5f + 0.5f * LOG2PI);
    }
    // next iteration's B1 orders phase-E global writes vs. new LDS traffic
  }
}

extern "C" void kernel_launch(void* const* d_in, const int* in_sizes, int n_in,
                              void* d_out, int out_size, void* d_ws, size_t ws_size,
                              hipStream_t stream) {
  (void)in_sizes; (void)n_in; (void)out_size; (void)ws_size;
  const float* inputs = (const float*)d_in[0];
  const float* W_ih   = (const float*)d_in[1];
  const float* W_hh   = (const float*)d_in[2];
  const float* b_ih   = (const float*)d_in[3];
  const float* b_hh   = (const float*)d_in[4];
  const float* W_mu   = (const float*)d_in[5];
  const float* b_mu   = (const float*)d_in[6];
  const float* W_sig  = (const float*)d_in[7];
  const float* b_sig  = (const float*)d_in[8];
  const float* eps    = (const float*)d_in[9];
  // d_in[10] = num_samples (hardcoded SS=4)

  __bf16* Wg  = (__bf16*)d_ws;                 // 2048*832 bf16  (3.41 MB)
  __bf16* Wl  = Wg + WG_ELEMS;                 // 128*512 bf16   (0.13 MB)
  float*  bia = (float*)(Wl + WL_ELEMS);       // 2048 f32

  vrnn_prep_kernel<<<512, 256, 0, stream>>>(W_ih, W_hh, b_ih, b_hh, W_mu, W_sig, Wg, Wl, bia);
  vrnn_kernel<<<NN / 16, 512, 0, stream>>>(inputs, eps, b_mu, b_sig, Wg, Wl, bia, (float*)d_out);
}
